// EngineeringGNN_1305670058364
// MI455X (gfx1250) — matrix operand order repaired and checked by reading, not compiled, not run on hardware
//
#include <hip/hip_runtime.h>
#include <hip/hip_bf16.h>

// ---------------------------------------------------------------------------
// EngineeringGNN forward for MI455X (gfx1250, wave32).
//  * GEMMs: v_wmma_f32_16x16x32_f16, f16 operands / f32 accum.
//  * Edge embedding recomputed per layer fused with message passing (never
//    materialized; only L2-resident buffers stream).
//  * A-tile staging via Tensor Data Mover (tensor_load_to_lds + tensorcnt).
//  * Scatter-sum via packed GLOBAL_ATOMIC_PK_ADD_F16 (half the atomic ops).
// ---------------------------------------------------------------------------

#define NNODES 50000
#define NEDGES 800000
#define YIELDV 250000000.0f

typedef __attribute__((ext_vector_type(16))) _Float16 v16h;
typedef __attribute__((ext_vector_type(2)))  __fp16   v2fp16;
typedef __attribute__((ext_vector_type(8)))  float    v8f;
typedef __attribute__((ext_vector_type(4)))  unsigned u32x4;
typedef __attribute__((ext_vector_type(8)))  int      i32x8;
typedef __attribute__((ext_vector_type(4)))  int      i32x4;

union FragH { v16h v; unsigned u[8]; };
union PkH   { v2fp16 h; unsigned u; };

// ---- input indices: setup_inputs() dict order, then params pytree leaves
enum {
  IN_X = 0, IN_EA, IN_EIDX, IN_BATCH, IN_POS, IN_FORCE, IN_MAT, IN_SECT,
  P_C0L1W, P_C0L1B, P_C0L2W, P_C0L2B,
  P_C1L1W, P_C1L1B, P_C1L2W, P_C1L2B,
  P_C2L1W, P_C2L1B, P_C2L2W, P_C2L2B,
  P_DL1W, P_DL1B, P_DL2W, P_DL2B,
  P_EL1W, P_EL1B, P_EL2W, P_EL2B, P_ELNB, P_ELNG,
  P_LOGBASE,
  P_SL1W, P_SL1B, P_SL2W, P_SL2B,
  P_NL1W, P_NL1B, P_NL2W, P_NL2B, P_NLNB, P_NLNG,
  P_PN0B, P_PN0G, P_PN1B, P_PN1G, P_PN2B, P_PN2G,
  P_SML1W, P_SML1B, P_SML2W, P_SML2B, P_SML3W, P_SML3B, P_SMLNB, P_SMLNG
};

// ---- output layout (floats), tuple flattened in return order
#define O_U     0
#define O_RAWU  150000
#define O_S     300000
#define O_LOGS  350000
#define O_MEAN  400000
#define O_DSG   400001
#define O_SAFE  400009

// ---- workspace layout (bytes)
#define WS_H      0u                                // 50000*128 f32
#define WS_AGGR   (WS_H + 50000u*128u*4u)           // 50000*128 f16
#define WS_RAWU   (WS_AGGR + 50000u*128u*2u)        // 50000*3 f32
#define WS_STATS  (WS_RAWU + 50000u*3u*4u)
#define WS_W16    (WS_STATS + 256u)
// f16 weight pool offsets (elements)
#define OW_NODE1  0
#define OW_NODE2  4096
#define OW_EDGE1  20480
#define OW_EDGE2  24576
#define OW_CONV(l) (40960 + (l)*32768)
#define OW_DISP1  139264
#define OW_STR1   147456

// ---------------------------------------------------------------------------
// WMMA fragment helpers (wave32); layouts per ISA 7.12.2.
// ---------------------------------------------------------------------------
__device__ __forceinline__ int frag_k_a(int j, int lane) {
  return ((j >> 2) << 4) + ((lane & 16) >> 1) + ((j & 3) << 1);
}
__device__ __forceinline__ int frag_k_b(int j, int lane) {
  return ((j >> 2) << 4) + ((lane & 16) >> 1) + ((j & 3) << 1);
}
__device__ __forceinline__ v16h load_frag_a(const _Float16* rowbase, int k0, int lane) {
  FragH f;
#pragma unroll
  for (int j = 0; j < 8; ++j)
    f.u[j] = *(const unsigned*)(rowbase + k0 + frag_k_a(j, lane));
  return f.v;
}
__device__ __forceinline__ v16h load_frag_b(const _Float16* rowbase, int k0, int lane) {
  FragH f;
#pragma unroll
  for (int j = 0; j < 8; ++j)
    f.u[j] = *(const unsigned*)(rowbase + k0 + frag_k_b(j, lane));
  return f.v;
}
__device__ __forceinline__ v8f wmma16(v16h a, v16h b, v8f c) {
  return __builtin_amdgcn_wmma_f32_16x16x32_f16(false, a, false, b, (short)0, c,
                                                false, false);
}
__device__ __forceinline__ float wave_sum32(float v) {
#pragma unroll
  for (int m = 16; m >= 1; m >>= 1) v += __shfl_xor(v, m, 32);
  return v;
}

// ---------------------------------------------------------------------------
// Tensor Data Mover: 1-row 2D tile (contiguous copy) global -> LDS.
// D# built per ISA 8.3/8.4: group0 {count=1, lds_addr, global_addr, type=2},
// group1 {data_size, tensor_dim0 (OOB zero-fill), tensor_dim1=1, tile_dim0,
// tile_dim1=1, stride0}.  sizeCode: 1 = 2B elems, 2 = 4B elems.
// Issue from ONE wave only; completion via s_wait_tensorcnt.
// ---------------------------------------------------------------------------
__device__ __forceinline__ void tdm_load_1d(unsigned lds_off, const void* g,
                                            unsigned tensorElems, unsigned tileElems,
                                            unsigned sizeCode) {
  unsigned long long ga = (unsigned long long)g;
  u32x4 g0;
  g0[0] = 1u;                                           // count=1 (valid), user mode
  g0[1] = lds_off;                                      // LDS byte address
  g0[2] = (unsigned)ga;                                 // global_addr[31:0]
  g0[3] = (unsigned)((ga >> 32) & 0x01ffffffu) | 0x80000000u;  // addr[56:32] | type=2
  i32x8 g1;
  g1[0] = (int)(sizeCode << 16);                        // wg_mask=0, data_size
  g1[1] = (int)((tensorElems & 0xffffu) << 16);         // tensor_dim0 lo
  g1[2] = (int)((tensorElems >> 16) & 0xffffu) | (1 << 16);   // dim0 hi | tensor_dim1=1
  g1[3] = (int)((tileElems & 0xffffu) << 16);           // tile_dim0
  g1[4] = 1;                                            // tile_dim1=1, tile_dim2=0
  g1[5] = (int)tileElems;                               // tensor_dim0_stride lo
  g1[6] = 0;
  g1[7] = 0;
  i32x4 z4 = {0, 0, 0, 0};
  i32x8 z8 = {0, 0, 0, 0, 0, 0, 0, 0};
  __builtin_amdgcn_tensor_load_to_lds(g0, g1, z4, z4, z8, 0);
}

// packed no-return f16 atomic add (GLOBAL_ATOMIC_PK_ADD_F16)
__device__ __forceinline__ void atomic_pk_add_f16(_Float16* p, float lo, float hi) {
  PkH cv; cv.h = __builtin_amdgcn_cvt_pkrtz(lo, hi);
  unsigned long long a = (unsigned long long)p;
  asm volatile("global_atomic_pk_add_f16 %0, %1, off" :: "v"(a), "v"(cv.u) : "memory");
}

// ---------------------------------------------------------------------------
// utility kernels
// ---------------------------------------------------------------------------
__global__ void k_zero4(float4* p, int n4) {
  int i = blockIdx.x * blockDim.x + threadIdx.x;
  if (i < n4) p[i] = float4{0.f, 0.f, 0.f, 0.f};
}
__global__ void k_init_stats(float* stats) {
  int i = threadIdx.x;
  if (i < 8) {
    stats[i] = 0.f;
    stats[8 + i] = 0.f;
    ((int*)stats)[16 + i] = 0x7f7fffff;   // zmin bits (FLT_MAX, z >= 0)
    ((int*)stats)[24 + i] = 0;            // zmax bits
  }
}
// W (KxN f32) -> Wt (N x Kpad f16, zero padded, transposed)
__global__ void k_convert_wt(const float* __restrict__ W, _Float16* __restrict__ Wt,
                             int K, int N, int Kpad) {
  int i = blockIdx.x * blockDim.x + threadIdx.x;
  if (i >= N * Kpad) return;
  int n = i / Kpad, k = i - n * Kpad;
  Wt[i] = (k < K) ? (_Float16)W[k * N + n] : (_Float16)0.f;
}

// ---------------------------------------------------------------------------
// Fused MLP2 (+LN), 32 rows / block, 8 waves.
// MODE 0: H = LN(l2(relu(l1(X))))                        (node encoder)
// MODE 1: H = LN(X + relu(l2(relu(l1(X + Add_f16)))))    (conv, in place)
// ---------------------------------------------------------------------------
template <int KPAD, int MODE>
__global__ __launch_bounds__(256)
void k_fused_mlp2(const float* __restrict__ X, const _Float16* __restrict__ Add,
                  float* __restrict__ Hout, int M, int Kin,
                  const _Float16* __restrict__ W1t, const float* __restrict__ b1,
                  const _Float16* __restrict__ W2t, const float* __restrict__ b2,
                  const float* __restrict__ lng, const float* __restrict__ lnb) {
  __shared__ __align__(16) float    Xs[32 * KPAD];
  __shared__ __align__(16) _Float16 AggrS[(MODE == 1) ? 32 * 128 : 8];
  __shared__ __align__(16) _Float16 At[32 * KPAD];
  __shared__ __align__(16) _Float16 T1[32 * 128];
  __shared__ __align__(16) float    Ot[32 * 128];
  const int tid = threadIdx.x, lane = tid & 31, wid = tid >> 5;
  const int row0 = blockIdx.x * 32;
  const int rem = M - row0;

  if (wid == 0) {  // TDM staging (wave 0 issues; OOB tail zero-filled via tensor_dim)
    unsigned tile = 32u * (unsigned)Kin;
    unsigned tens = (rem >= 32) ? tile : (unsigned)(rem * Kin);
    tdm_load_1d((unsigned)(size_t)Xs, X + (size_t)row0 * Kin, tens, tile, 2u);
    if (MODE == 1) {
      unsigned tens2 = (rem >= 32) ? 4096u : (unsigned)(rem * 128);
      tdm_load_1d((unsigned)(size_t)AggrS, Add + (size_t)row0 * 128, tens2, 4096u, 1u);
    }
    __builtin_amdgcn_s_wait_tensorcnt(0);
  }
  __syncthreads();

  // convert staged f32 (+f16 aggr) -> f16 A tile, packed pair stores
  for (int i = tid * 2; i < 32 * KPAD; i += 512) {
    int r = i / KPAD, k = i - (i / KPAD) * KPAD;
    float v0 = 0.f, v1 = 0.f;
    if (k < Kin) {
      v0 = Xs[r * Kin + k];
      if (MODE == 1) v0 += (float)AggrS[r * 128 + k];
    }
    if (k + 1 < Kin) {
      v1 = Xs[r * Kin + k + 1];
      if (MODE == 1) v1 += (float)AggrS[r * 128 + k + 1];
    }
    PkH cv; cv.h = __builtin_amdgcn_cvt_pkrtz(v0, v1);
    *(unsigned*)&At[i] = cv.u;
  }
  __syncthreads();

  // GEMM1: (32 x KPAD) x (KPAD x 128) -> T1 (relu + b1)
  {
    const int n = wid * 16 + (lane & 15);
    v16h fb[KPAD / 32];
#pragma unroll
    for (int ks = 0; ks < KPAD / 32; ++ks)
      fb[ks] = load_frag_b(W1t + n * KPAD, ks * 32, lane);
    const float bv = b1[n];
#pragma unroll
    for (int rt = 0; rt < 2; ++rt) {
      const _Float16* arow = At + (rt * 16 + (lane & 15)) * KPAD;
      v8f acc = {};
#pragma unroll
      for (int ks = 0; ks < KPAD / 32; ++ks)
        acc = wmma16(load_frag_a(arow, ks * 32, lane), fb[ks], acc);
#pragma unroll
      for (int r = 0; r < 8; ++r) {
        int m = rt * 16 + r + ((lane & 16) >> 1);
        float v = acc[r] + bv;
        T1[m * 128 + n] = (_Float16)(v > 0.f ? v : 0.f);
      }
    }
  }
  __syncthreads();

  // GEMM2: (32 x 128) x (128 x 128) -> Ot (+b2)
  {
    const int n = wid * 16 + (lane & 15);
    v16h fb[4];
#pragma unroll
    for (int ks = 0; ks < 4; ++ks)
      fb[ks] = load_frag_b(W2t + n * 128, ks * 32, lane);
    const float bv = b2[n];
#pragma unroll
    for (int rt = 0; rt < 2; ++rt) {
      const _Float16* arow = T1 + (rt * 16 + (lane & 15)) * 128;
      v8f acc = {};
#pragma unroll
      for (int ks = 0; ks < 4; ++ks)
        acc = wmma16(load_frag_a(arow, ks * 32, lane), fb[ks], acc);
#pragma unroll
      for (int r = 0; r < 8; ++r)
        Ot[(rt * 16 + r + ((lane & 16) >> 1)) * 128 + n] = acc[r] + bv;
    }
  }
  __syncthreads();

  // (residual + relu +) LayerNorm; wave handles 4 rows, lane covers 4 cols
#pragma unroll
  for (int i = 0; i < 4; ++i) {
    const int r = wid * 4 + i, g = row0 + r;
    float v[4];
#pragma unroll
    for (int c = 0; c < 4; ++c) {
      float o = Ot[r * 128 + lane + 32 * c];
      if (MODE == 1) {
        float hres = Xs[r * 128 + lane + 32 * c];  // staged h (zero-filled OOB)
        v[c] = hres + (o > 0.f ? o : 0.f);
      } else {
        v[c] = o;
      }
    }
    float mean = wave_sum32(v[0] + v[1] + v[2] + v[3]) * (1.f / 128.f);
    float d0 = v[0] - mean, d1 = v[1] - mean, d2 = v[2] - mean, d3 = v[3] - mean;
    float var = wave_sum32(d0 * d0 + d1 * d1 + d2 * d2 + d3 * d3) * (1.f / 128.f);
    float rs = rsqrtf(var + 1e-5f);
    if (g < M) {
      Hout[g * 128 + lane]      = d0 * rs * lng[lane]      + lnb[lane];
      Hout[g * 128 + lane + 32] = d1 * rs * lng[lane + 32] + lnb[lane + 32];
      Hout[g * 128 + lane + 64] = d2 * rs * lng[lane + 64] + lnb[lane + 64];
      Hout[g * 128 + lane + 96] = d3 * rs * lng[lane + 96] + lnb[lane + 96];
    }
  }
}

// ---------------------------------------------------------------------------
// Fused edge pipeline, 32 edges / block: recompute edge-encoder MLP2+LN in
// LDS, then msg = relu(h[src]+ea) scattered with packed f16 atomics.
// ---------------------------------------------------------------------------
__global__ __launch_bounds__(256)
void k_edge_msg(const float* __restrict__ EA, const int* __restrict__ src,
                const int* __restrict__ dst, const float* __restrict__ h,
                _Float16* __restrict__ aggr,
                const _Float16* __restrict__ W1t, const float* __restrict__ b1,
                const _Float16* __restrict__ W2t, const float* __restrict__ b2,
                const float* __restrict__ lng, const float* __restrict__ lnb) {
  __shared__ __align__(16) float    Eas[32 * 6];
  __shared__ __align__(16) _Float16 At[32 * 32];
  __shared__ __align__(16) _Float16 T1[32 * 128];
  __shared__ __align__(16) float    Ot[32 * 128];
  __shared__ int sSrc[32], sDst[32];
  const int tid = threadIdx.x, lane = tid & 31, wid = tid >> 5;
  const int e0 = blockIdx.x * 32;

  if (wid == 0) {
    tdm_load_1d((unsigned)(size_t)Eas, EA + (size_t)e0 * 6, 192u, 192u, 2u);
    __builtin_amdgcn_s_wait_tensorcnt(0);
  }
  if (tid >= 32 && tid < 64) {
    int t = tid - 32, e = e0 + t;
    sSrc[t] = (e < NEDGES) ? src[e] : 0;
    sDst[t] = (e < NEDGES) ? dst[e] : 0;
  }
  __syncthreads();

  for (int i = tid * 2; i < 32 * 32; i += 512) {
    int r = i >> 5, k = i & 31;
    float v0 = (k < 6) ? Eas[r * 6 + k] : 0.f;
    float v1 = (k + 1 < 6) ? Eas[r * 6 + k + 1] : 0.f;
    PkH cv; cv.h = __builtin_amdgcn_cvt_pkrtz(v0, v1);
    *(unsigned*)&At[i] = cv.u;
  }
  __syncthreads();

  { // GEMM1: (32x32) x (32x128)
    const int n = wid * 16 + (lane & 15);
    const v16h fb = load_frag_b(W1t + n * 32, 0, lane);
    const float bv = b1[n];
#pragma unroll
    for (int rt = 0; rt < 2; ++rt) {
      const _Float16* arow = At + (rt * 16 + (lane & 15)) * 32;
      v8f acc = {};
      acc = wmma16(load_frag_a(arow, 0, lane), fb, acc);
#pragma unroll
      for (int r = 0; r < 8; ++r) {
        int m = rt * 16 + r + ((lane & 16) >> 1);
        float v = acc[r] + bv;
        T1[m * 128 + n] = (_Float16)(v > 0.f ? v : 0.f);
      }
    }
  }
  __syncthreads();

  { // GEMM2: (32x128) x (128x128)
    const int n = wid * 16 + (lane & 15);
    v16h fb[4];
#pragma unroll
    for (int ks = 0; ks < 4; ++ks)
      fb[ks] = load_frag_b(W2t + n * 128, ks * 32, lane);
    const float bv = b2[n];
#pragma unroll
    for (int rt = 0; rt < 2; ++rt) {
      const _Float16* arow = T1 + (rt * 16 + (lane & 15)) * 128;
      v8f acc = {};
#pragma unroll
      for (int ks = 0; ks < 4; ++ks)
        acc = wmma16(load_frag_a(arow, ks * 32, lane), fb[ks], acc);
#pragma unroll
      for (int r = 0; r < 8; ++r)
        Ot[(rt * 16 + r + ((lane & 16) >> 1)) * 128 + n] = acc[r] + bv;
    }
  }
  __syncthreads();

  // LayerNorm rows in LDS
#pragma unroll
  for (int i = 0; i < 4; ++i) {
    const int r = wid * 4 + i;
    float v0 = Ot[r * 128 + lane], v1 = Ot[r * 128 + lane + 32];
    float v2 = Ot[r * 128 + lane + 64], v3 = Ot[r * 128 + lane + 96];
    float mean = wave_sum32(v0 + v1 + v2 + v3) * (1.f / 128.f);
    float d0 = v0 - mean, d1 = v1 - mean, d2 = v2 - mean, d3 = v3 - mean;
    float var = wave_sum32(d0 * d0 + d1 * d1 + d2 * d2 + d3 * d3) * (1.f / 128.f);
    float rs = rsqrtf(var + 1e-5f);
    Ot[r * 128 + lane]      = d0 * rs * lng[lane]      + lnb[lane];
    Ot[r * 128 + lane + 32] = d1 * rs * lng[lane + 32] + lnb[lane + 32];
    Ot[r * 128 + lane + 64] = d2 * rs * lng[lane + 64] + lnb[lane + 64];
    Ot[r * 128 + lane + 96] = d3 * rs * lng[lane + 96] + lnb[lane + 96];
  }
  __syncthreads();

  // message: wave handles 4 edges; lane covers cols 4*lane..4*lane+3
#pragma unroll
  for (int i = 0; i < 4; ++i) {
    const int el = wid * 4 + i, e = e0 + el;
    if (e >= NEDGES) continue;
    const int s = sSrc[el], d = sDst[el];
    const float4 hv = ((const float4*)(h + (size_t)s * 128))[lane];
    const float4 ev = ((const float4*)(Ot + el * 128))[lane];
    _Float16* ab = aggr + (size_t)d * 128 + lane * 4;
    atomic_pk_add_f16(ab,     fmaxf(hv.x + ev.x, 0.f), fmaxf(hv.y + ev.y, 0.f));
    atomic_pk_add_f16(ab + 2, fmaxf(hv.z + ev.z, 0.f), fmaxf(hv.w + ev.w, 0.f));
  }
}

// ---------------------------------------------------------------------------
// heads: disp_head (128->64 relu ->3) + log_stress_head (128->64 relu ->1)
// ---------------------------------------------------------------------------
__global__ __launch_bounds__(256)
void k_heads(const float* __restrict__ h, int M,
             const _Float16* __restrict__ Wd1t, const float* __restrict__ bd1,
             const float* __restrict__ Wd2, const float* __restrict__ bd2,
             const _Float16* __restrict__ Ws1t, const float* __restrict__ bs1,
             const float* __restrict__ Ws2, const float* __restrict__ bs2,
             float* __restrict__ raw_un, float* __restrict__ dout) {
  __shared__ __align__(16) float    Xs[32 * 128];
  __shared__ __align__(16) _Float16 At[32 * 128];
  __shared__ __align__(16) _Float16 T1d[32 * 64];
  __shared__ __align__(16) _Float16 T1s[32 * 64];
  const int tid = threadIdx.x, lane = tid & 31, wid = tid >> 5;
  const int row0 = blockIdx.x * 32;
  const int rem = M - row0;

  if (wid == 0) {
    unsigned tens = (rem >= 32) ? 4096u : (unsigned)(rem * 128);
    tdm_load_1d((unsigned)(size_t)Xs, h + (size_t)row0 * 128, tens, 4096u, 2u);
    __builtin_amdgcn_s_wait_tensorcnt(0);
  }
  __syncthreads();
  for (int i = tid * 2; i < 32 * 128; i += 512) {
    PkH cv; cv.h = __builtin_amdgcn_cvt_pkrtz(Xs[i], Xs[i + 1]);
    *(unsigned*)&At[i] = cv.u;
  }
  __syncthreads();

  { // 8 waves cover 2 rowtiles x 4 coltiles of 32x64; A frags shared by heads
    const int rt = wid >> 2, n = (wid & 3) * 16 + (lane & 15);
    const _Float16* arow = At + (rt * 16 + (lane & 15)) * 128;
    v16h fa[4];
#pragma unroll
    for (int ks = 0; ks < 4; ++ks) fa[ks] = load_frag_a(arow, ks * 32, lane);
    {
      v8f acc = {};
#pragma unroll
      for (int ks = 0; ks < 4; ++ks)
        acc = wmma16(fa[ks], load_frag_b(Wd1t + n * 128, ks * 32, lane), acc);
      const float bv = bd1[n];
#pragma unroll
      for (int r = 0; r < 8; ++r) {
        int m = rt * 16 + r + ((lane & 16) >> 1);
        float v = acc[r] + bv;
        T1d[m * 64 + n] = (_Float16)(v > 0.f ? v : 0.f);
      }
    }
    {
      v8f acc = {};
#pragma unroll
      for (int ks = 0; ks < 4; ++ks)
        acc = wmma16(fa[ks], load_frag_b(Ws1t + n * 128, ks * 32, lane), acc);
      const float bv = bs1[n];
#pragma unroll
      for (int r = 0; r < 8; ++r) {
        int m = rt * 16 + r + ((lane & 16) >> 1);
        float v = acc[r] + bv;
        T1s[m * 64 + n] = (_Float16)(v > 0.f ? v : 0.f);
      }
    }
  }
  __syncthreads();

  const int row = tid >> 3, c = tid & 7, g = row0 + row;
  if (g < M) {
    if (c < 3) {
      float s = bd2[c];
#pragma unroll 8
      for (int k = 0; k < 64; ++k) s += (float)T1d[row * 64 + k] * Wd2[k * 3 + c];
      raw_un[g * 3 + c] = s;
    } else if (c == 3) {
      float s = bs2[0];
#pragma unroll 8
      for (int k = 0; k < 64; ++k) s += (float)T1s[row * 64 + k] * Ws2[k];
      float ls = fminf(fmaxf(s, 0.f), 30.f);
      float se = expf(ls);
      dout[O_LOGS + g] = ls;
      dout[O_S + g] = se;
      dout[O_SAFE + g] = YIELDV / (se + 1e-8f);
    }
  }
}

// ---------------------------------------------------------------------------
__global__ void k_batch_stats(const float* __restrict__ raw_un,
                              const int* __restrict__ batch,
                              const float* __restrict__ pos, float* stats) {
  int n = blockIdx.x * blockDim.x + threadIdx.x;
  if (n >= NNODES) return;
  int b = batch[n];
  float x = raw_un[n * 3 + 0], y = raw_un[n * 3 + 1], z3 = raw_un[n * 3 + 2];
  atomicAdd(&stats[b], 1.0f);
  atomicAdd(&stats[8 + b], x * x + y * y + z3 * z3);
  float z = pos[n * 3 + 2];
  atomicMin(((int*)stats) + 16 + b, __float_as_int(z));
  atomicMax(((int*)stats) + 24 + b, __float_as_int(z));
}

__global__ void k_scale(const float* __restrict__ force, const float* __restrict__ mat,
                        const float* __restrict__ sect,
                        const float* __restrict__ W1, const float* __restrict__ b1,
                        const float* __restrict__ W2, const float* __restrict__ b2,
                        const float* __restrict__ W3, const float* __restrict__ b3,
                        const float* __restrict__ lnb, const float* __restrict__ lng,
                        const float* __restrict__ log_base,
                        float* stats, float* dout) {
  __shared__ float feats[8][6];
  __shared__ float t1[128];
  __shared__ float t2[64];
  __shared__ float mstat[2];
  const int tid = threadIdx.x;
  if (tid < 8) {
    float fx = force[tid * 3], fy = force[tid * 3 + 1], fz = force[tid * 3 + 2];
    float logF = logf(sqrtf(fx * fx + fy * fy + fz * fz) + 1.f);
    float E = fmaxf(mat[tid * 2], 1.f);
    float logE = logf(E + 1e-12f);
    float nu = fminf(fmaxf(mat[tid * 2 + 1], 0.f), 0.49f);
    float zmn = __int_as_float(((int*)stats)[16 + tid]);
    float zmx = __int_as_float(((int*)stats)[24 + tid]);
    float L = fmaxf(zmx - zmn, 1e-6f);
    float logL = logf(L + 1e-6f);
    float I = fmaxf(sect[tid * 6 + 3], 1e-18f);
    float logI = logf(I + 1e-18f);
    feats[tid][0] = logF; feats[tid][1] = logE; feats[tid][2] = nu;
    feats[tid][3] = logL; feats[tid][4] = logI;
    feats[tid][5] = logF + 3.f * logL - logE - logI;
  }
  __syncthreads();
  const float base = 1e-6f + log1pf(expf(log_base[0]));
  float meanacc = 0.f;
  for (int b = 0; b < 8; ++b) {
    {
      float s = b1[tid];
#pragma unroll
      for (int i = 0; i < 6; ++i) s += feats[b][i] * W1[i * 128 + tid];
      t1[tid] = s > 0.f ? s : 0.f;
    }
    __syncthreads();
    if (tid < 64) {
      float s = b2[tid];
      for (int k = 0; k < 128; ++k) s += t1[k] * W2[k * 64 + tid];
      t2[tid] = s;
    }
    __syncthreads();
    if (tid == 0) {
      float m = 0.f;
      for (int k = 0; k < 64; ++k) m += t2[k];
      m *= (1.f / 64.f);
      float v = 0.f;
      for (int k = 0; k < 64; ++k) { float d = t2[k] - m; v += d * d; }
      mstat[0] = m; mstat[1] = rsqrtf(v * (1.f / 64.f) + 1e-5f);
    }
    __syncthreads();
    if (tid < 64) {
      float z = (t2[tid] - mstat[0]) * mstat[1] * lng[tid] + lnb[tid];
      t2[tid] = z > 0.f ? z : 0.f;
    }
    __syncthreads();
    if (tid == 0) {
      float lm = b3[0];
      for (int k = 0; k < 64; ++k) lm += t2[k] * W3[k];
      lm = 4.0f * tanhf(lm * 0.25f);
      float sc = fmaxf(base * expf(lm), 1e-6f);
      stats[32 + b] = sc;
      dout[O_DSG + b] = sc;
      meanacc += sc;
    }
    __syncthreads();
  }
  if (tid == 0) dout[O_MEAN] = meanacc * (1.f / 8.f);
}

__global__ void k_finalize(const float* __restrict__ raw_un,
                           const int* __restrict__ batch,
                           const float* __restrict__ stats, float* dout) {
  int n = blockIdx.x * blockDim.x + threadIdx.x;
  if (n >= NNODES) return;
  int b = batch[n];
  float cnt = fmaxf(stats[b], 1.f);
  float rms = fmaxf(sqrtf(stats[8 + b] / cnt), 1e-8f);
  float sc = stats[32 + b];
#pragma unroll
  for (int c = 0; c < 3; ++c) {
    float ru = raw_un[n * 3 + c] / rms;
    dout[O_RAWU + n * 3 + c] = ru;
    dout[O_U + n * 3 + c] = ru * sc;
  }
}

// ---------------------------------------------------------------------------
extern "C" void kernel_launch(void* const* d_in, const int* in_sizes, int n_in,
                              void* d_out, int out_size, void* d_ws, size_t ws_size,
                              hipStream_t stream) {
  (void)in_sizes; (void)n_in; (void)out_size; (void)ws_size;
  #define FP(i) ((const float*)d_in[i])
  const int* ei    = (const int*)d_in[IN_EIDX];
  const int* src   = ei;
  const int* dst   = ei + NEDGES;
  const int* batch = (const int*)d_in[IN_BATCH];

  char* ws = (char*)d_ws;
  float*     h      = (float*)(ws + WS_H);
  _Float16*  aggr   = (_Float16*)(ws + WS_AGGR);
  float*     raw_un = (float*)(ws + WS_RAWU);
  float*     stats  = (float*)(ws + WS_STATS);
  _Float16*  w16    = (_Float16*)(ws + WS_W16);
  float*     dout   = (float*)d_out;

  k_init_stats<<<1, 32, 0, stream>>>(stats);
  auto conv = [&](int pidx, _Float16* wdst, int K, int N, int Kpad) {
    int elems = N * Kpad;
    k_convert_wt<<<(elems + 255) / 256, 256, 0, stream>>>(FP(pidx), wdst, K, N, Kpad);
  };
  conv(P_NL1W, w16 + OW_NODE1, 12, 128, 32);
  conv(P_NL2W, w16 + OW_NODE2, 128, 128, 128);
  conv(P_EL1W, w16 + OW_EDGE1, 6, 128, 32);
  conv(P_EL2W, w16 + OW_EDGE2, 128, 128, 128);
  const int convW[3][2] = {{P_C0L1W, P_C0L2W}, {P_C1L1W, P_C1L2W}, {P_C2L1W, P_C2L2W}};
  for (int l = 0; l < 3; ++l) {
    conv(convW[l][0], w16 + OW_CONV(l), 128, 128, 128);
    conv(convW[l][1], w16 + OW_CONV(l) + 16384, 128, 128, 128);
  }
  conv(P_DL1W, w16 + OW_DISP1, 128, 64, 128);
  conv(P_SL1W, w16 + OW_STR1, 128, 64, 128);

  const int NODE_BLKS = (NNODES + 31) / 32;
  const int EDGE_BLKS = NEDGES / 32;

  k_fused_mlp2<32, 0><<<NODE_BLKS, 256, 0, stream>>>(
      FP(IN_X), nullptr, h, NNODES, 12,
      w16 + OW_NODE1, FP(P_NL1B), w16 + OW_NODE2, FP(P_NL2B),
      FP(P_NLNG), FP(P_NLNB));

  const int pnB[3] = {P_PN0B, P_PN1B, P_PN2B};
  const int pnG[3] = {P_PN0G, P_PN1G, P_PN2G};
  const int convB[3][2] = {{P_C0L1B, P_C0L2B}, {P_C1L1B, P_C1L2B}, {P_C2L1B, P_C2L2B}};
  for (int l = 0; l < 3; ++l) {
    // zero f16 aggr (12.8MB): 800000 float4s
    k_zero4<<<(800000 + 255) / 256, 256, 0, stream>>>((float4*)aggr, 800000);
    k_edge_msg<<<EDGE_BLKS, 256, 0, stream>>>(
        FP(IN_EA), src, dst, h, aggr,
        w16 + OW_EDGE1, FP(P_EL1B), w16 + OW_EDGE2, FP(P_EL2B),
        FP(P_ELNG), FP(P_ELNB));
    k_fused_mlp2<128, 1><<<NODE_BLKS, 256, 0, stream>>>(
        h, aggr, h, NNODES, 128,
        w16 + OW_CONV(l), FP(convB[l][0]), w16 + OW_CONV(l) + 16384, FP(convB[l][1]),
        FP(pnG[l]), FP(pnB[l]));
  }

  k_heads<<<NODE_BLKS, 256, 0, stream>>>(
      h, NNODES,
      w16 + OW_DISP1, FP(P_DL1B), FP(P_DL2W), FP(P_DL2B),
      w16 + OW_STR1, FP(P_SL1B), FP(P_SL2W), FP(P_SL2B),
      raw_un, dout);

  k_batch_stats<<<(NNODES + 255) / 256, 256, 0, stream>>>(raw_un, batch, FP(IN_POS), stats);
  k_scale<<<1, 128, 0, stream>>>(
      FP(IN_FORCE), FP(IN_MAT), FP(IN_SECT),
      FP(P_SML1W), FP(P_SML1B), FP(P_SML2W), FP(P_SML2B),
      FP(P_SML3W), FP(P_SML3B), FP(P_SMLNB), FP(P_SMLNG),
      FP(P_LOGBASE), stats, dout);
  k_finalize<<<(NNODES + 255) / 256, 256, 0, stream>>>(raw_un, batch, stats, dout);
  #undef FP
}
